// Eq1to2Net_26817775796962
// MI455X (gfx1250) — matrix operand order repaired and checked by reading, not compiled, not run on hardware
//
#include <hip/hip_runtime.h>
#include <hip/hip_bf16.h>

#define NN_   16
#define DIN_  32
#define DOUT_ 32
#define MM_   128

typedef __attribute__((ext_vector_type(16))) _Float16 v16h;
typedef __attribute__((ext_vector_type(8)))  _Float16 v8h;
typedef __attribute__((ext_vector_type(8)))  float    v8f;

union V16 { v16h v; v8h h[2]; };

// ---------------------------------------------------------------------------
// Kernel 1: h0 = relu(eq1to2(x, coefs0, bias0))
// out[n,s,i,j] = A[i] + B[j] + C + bias + delta_ij*(Ad[i] + Cd)
// grid (N, DOUT), 128 threads (thread = i)
// ---------------------------------------------------------------------------
__global__ void __launch_bounds__(128)
k_eq1to2(const float* __restrict__ x, const float* __restrict__ c0,
         const float* __restrict__ b0, float* __restrict__ h) {
  const int nn = blockIdx.x, s = blockIdx.y, i = threadIdx.x;
  __shared__ float Bs[MM_];
  __shared__ float Ssh;
  __shared__ float wr[4];
  float A = 0.f, Bv = 0.f, Ad = 0.f, C = 0.f, Cd = 0.f;
  const float* xn = x + (size_t)nn * DIN_ * MM_;
  for (int d = 0; d < DIN_; ++d) {
    float v = xn[d * MM_ + i];
    const float* cd = c0 + (d * DOUT_ + s) * 5;
    A  += cd[2] * v;   // b3: x[i] broadcast over cols
    Bv += cd[3] * v;   // b4: x[j] broadcast over rows
    Ad += cd[0] * v;   // b1: diag_embed(x)
    // block-reduce v -> S_d
    float r = v;
    for (int off = 16; off; off >>= 1) r += __shfl_xor(r, off);
    if ((threadIdx.x & 31) == 0) wr[threadIdx.x >> 5] = r;
    __syncthreads();
    if (threadIdx.x == 0) Ssh = wr[0] + wr[1] + wr[2] + wr[3];
    __syncthreads();
    float Sd = Ssh;
    C  += cd[4] * Sd;  // b5: sum everywhere
    Cd += cd[1] * Sd;  // b2: sum on diagonal
  }
  Bs[i] = Bv;
  __syncthreads();
  const float bias = b0[s];
  const float diagadd = Ad + Cd;
  float* ho = h + (((size_t)nn * DOUT_ + s) * MM_ + i) * MM_;
  for (int j = 0; j < MM_; ++j) {
    float v = A + Bs[j] + C + bias;
    if (j == i) v += diagadd;
    ho[j] = fmaxf(v, 0.f);
  }
}

// ---------------------------------------------------------------------------
// Kernel 2a: per-(n,d) reductions of h: rsum[128], csum[128], diag[128],
// tsum, sdiag  ->  red[n][d][512]
// grid (N, DOUT), 128 threads (4 waves)
// ---------------------------------------------------------------------------
__global__ void __launch_bounds__(128)
k_reduce(const float* __restrict__ h, float* __restrict__ red) {
  const int nn = blockIdx.x, d = blockIdx.y;
  const float* xd = h + ((size_t)nn * DOUT_ + d) * MM_ * MM_;
  float* rout = red + ((size_t)nn * DOUT_ + d) * 512;
  const int tid = threadIdx.x, w = tid >> 5, l = tid & 31;
  // row sums: wave w handles rows 32w..32w+31
  for (int rr = 0; rr < 32; ++rr) {
    int i = w * 32 + rr;
    const float* row = xd + i * MM_;
    float s = row[l] + row[l + 32] + row[l + 64] + row[l + 96];
    for (int off = 16; off; off >>= 1) s += __shfl_xor(s, off);
    if (l == 0) rout[i] = s;
  }
  // column sums: thread = column j
  float cs = 0.f;
  for (int i = 0; i < MM_; ++i) {
    if (i + 8 < MM_) __builtin_prefetch(xd + (i + 8) * MM_ + tid, 0, 1);
    cs += xd[i * MM_ + tid];
  }
  rout[128 + tid] = cs;
  float dg = xd[tid * MM_ + tid];
  rout[256 + tid] = dg;
  // tsum (=sum of csum) and sdiag (=sum of diag)
  __shared__ float wr[8];
  float t = cs, sd = dg;
  for (int off = 16; off; off >>= 1) { t += __shfl_xor(t, off); sd += __shfl_xor(sd, off); }
  if (l == 0) { wr[w] = t; wr[4 + w] = sd; }
  __syncthreads();
  if (tid == 0) rout[384] = wr[0] + wr[1] + wr[2] + wr[3];  // tsum
  if (tid == 1) rout[385] = wr[4] + wr[5] + wr[6] + wr[7];  // sdiag
}

// ---------------------------------------------------------------------------
// Kernel 2b: mix reductions over channels with coefs ->
// mix[n][s][512] = { colvec[128], rowvec[128], diagvec[128], scalar, dscalar }
// grid N, 128 threads (thread = i)
// ---------------------------------------------------------------------------
__global__ void __launch_bounds__(128)
k_mix(const float* __restrict__ red, const float* __restrict__ cf,
      float* __restrict__ mix) {
  const int nn = blockIdx.x, i = threadIdx.x;
  for (int s = 0; s < DOUT_; ++s) {
    float colv = 0.f, rowv = 0.f, dgv = 0.f;
    for (int d = 0; d < DOUT_; ++d) {
      const float* r = red + ((size_t)nn * DOUT_ + d) * 512;
      const float* c = cf + (d * DOUT_ + s) * 15;
      float rs = r[i], csm = r[128 + i], dg = r[256 + i];
      colv += c[5] * csm + c[6] * rs + c[11] * dg;   // b6,b7,b12 (i-broadcast)
      rowv += c[7] * csm + c[8] * rs + c[12] * dg;   // b8,b9,b13 (j-broadcast)
      dgv  += c[0] * dg  + c[2] * rs + c[3]  * csm;  // b1,b3,b4  (diagonal)
    }
    float* mo = mix + ((size_t)nn * DOUT_ + s) * 512;
    mo[i] = colv; mo[128 + i] = rowv; mo[256 + i] = dgv;
    if (i == 0) {
      float sc = 0.f, dsc = 0.f;
      for (int d = 0; d < DOUT_; ++d) {
        const float* r = red + ((size_t)nn * DOUT_ + d) * 512;
        const float* c = cf + (d * DOUT_ + s) * 15;
        float ts = r[384], sdg = r[385];
        sc  += c[13] * sdg + c[14] * ts;  // b14,b15 (everywhere)
        dsc += c[1]  * sdg + c[4]  * ts;  // b2,b5   (diagonal)
      }
      mo[384] = sc; mo[385] = dsc;
    }
  }
}

// ---------------------------------------------------------------------------
// Kernel 2c: main eq2to2 dense terms via WMMA f32_16x16x32_f16 (K = D = 32):
//   out[s,i,j] = relu( WMMA(W10, x[:,i,j]) + WMMA(W11, x[:,j,i])
//                      + colvec[s,i] + rowvec[s,j] + scal[s]
//                      + delta_ij*(diagvec[s,i] + dscal[s]) )
// grid (N, 8 i-tiles, 8 j-tiles), 256 threads (8 waves)
// ---------------------------------------------------------------------------
__global__ void __launch_bounds__(256)
k_eq2to2_main(const float* __restrict__ x, const float* __restrict__ cf,
              const float* __restrict__ mix, const float* __restrict__ bias,
              const float* __restrict__ dbias, float* __restrict__ out) {
  const int nn = blockIdx.x, I0 = blockIdx.y * 16, J0 = blockIdx.z * 16;
  const int tid = threadIdx.x, lane = tid & 31, w = tid >> 5;
  // t[i][j][d]: i stride 648 halves, j stride 40 halves (padded, 16B-aligned)
  __shared__ __attribute__((aligned(16))) _Float16 t1[16 * 648];
  __shared__ __attribute__((aligned(16))) _Float16 t2[16 * 648];
  __shared__ __attribute__((aligned(16))) _Float16 cw10[32 * 32];  // [s][d]
  __shared__ __attribute__((aligned(16))) _Float16 cw11[32 * 32];
  __shared__ float colv[32 * 16], rowv[32 * 16], dgv[32 * 16], scl[32], dsc[32];

  const float* xn = x + (size_t)nn * DOUT_ * MM_ * MM_;
  for (int k = tid; k < 32 * 32; k += 256) {
    int d = k >> 5, s = k & 31;
    const float* c = cf + (d * DOUT_ + s) * 15;
    cw10[s * 32 + d] = (_Float16)c[9];    // b10: identity
    cw11[s * 32 + d] = (_Float16)c[10];   // b11: transpose
  }
  for (int k = tid; k < 32 * 16; k += 256) {
    int s = k >> 4, r = k & 15;
    const float* mo = mix + ((size_t)nn * DOUT_ + s) * 512;
    colv[k] = mo[I0 + r];
    rowv[k] = mo[128 + J0 + r];
    dgv[k]  = mo[256 + I0 + r];
  }
  if (tid < 32) {
    const float* mo = mix + ((size_t)nn * DOUT_ + tid) * 512;
    scl[tid] = mo[384] + bias[tid];
    dsc[tid] = mo[385] + dbias[tid];
  }
  // t1[i][j][d] = x[d, I0+i, J0+j]  (global-contiguous in j)
  for (int it = 0; it < 32; ++it) {
    int idx = tid + it * 256;
    int j = idx & 15, i = (idx >> 4) & 15, d = idx >> 8;
    t1[i * 648 + j * 40 + d] = (_Float16)xn[(size_t)d * MM_ * MM_ + (I0 + i) * MM_ + (J0 + j)];
  }
  // t2[i][j][d] = x[d, J0+j, I0+i]  (global-contiguous in i)
  for (int it = 0; it < 32; ++it) {
    int idx = tid + it * 256;
    int i = idx & 15, j = (idx >> 4) & 15, d = idx >> 8;
    t2[i * 648 + j * 40 + d] = (_Float16)xn[(size_t)d * MM_ * MM_ + (J0 + j) * MM_ + (I0 + i)];
  }
  __syncthreads();

  const int stile = w & 1, wq = w >> 1;
  const int hi = lane >> 4, ln = lane & 15;
  // A fragments (16x32 f16): lane m = ln (s-rel); halves h -> d = 16*hi + h
  V16 a10, a11;
  {
    const _Float16* b10 = &cw10[(stile * 16 + ln) * 32];
    const _Float16* b11 = &cw11[(stile * 16 + ln) * 32];
    a10.h[0] = *(const v8h*)(b10 + 16 * hi);
    a10.h[1] = *(const v8h*)(b10 + 16 * hi + 8);
    a11.h[0] = *(const v8h*)(b11 + 16 * hi);
    a11.h[1] = *(const v8h*)(b11 + 16 * hi + 8);
  }
  float* o = out + (size_t)nn * DOUT_ * MM_ * MM_;
  for (int t = 0; t < 4; ++t) {
    int irel = wq * 4 + t;
    // B fragments (32x16 f16): lane n = ln (j-rel); halves h -> d = 16*hi + h
    V16 b1, b2;
    const _Float16* p1 = &t1[irel * 648 + ln * 40 + 16 * hi];
    const _Float16* p2 = &t2[irel * 648 + ln * 40 + 16 * hi];
    b1.h[0] = *(const v8h*)p1;  b1.h[1] = *(const v8h*)(p1 + 8);
    b2.h[0] = *(const v8h*)p2;  b2.h[1] = *(const v8h*)(p2 + 8);
    v8f acc = {};
    acc = __builtin_amdgcn_wmma_f32_16x16x32_f16(false, a10.v, false, b1.v, (short)0, acc, false, false);
    acc = __builtin_amdgcn_wmma_f32_16x16x32_f16(false, a11.v, false, b2.v, (short)0, acc, false, false);
    const int jg = J0 + ln, ig = I0 + irel;
#pragma unroll
    for (int r = 0; r < 8; ++r) {
      int s = stile * 16 + r + 8 * hi;  // C/D layout: VGPR r -> M = r + 8*(lane>=16)
      float v = acc[r] + colv[s * 16 + irel] + rowv[s * 16 + ln] + scl[s];
      if (ig == jg) v += dgv[s * 16 + irel] + dsc[s];
      o[(size_t)s * MM_ * MM_ + ig * MM_ + jg] = fmaxf(v, 0.f);
    }
  }
}

// ---------------------------------------------------------------------------
// Kernel 4: fc  out[n,s,i,o] = sum_j h[n,s,i,j]*fw[o,j] + fb[o]  via WMMA
// grid (N, DOUT), 256 threads (8 waves; wave = 16-row i-tile)
// ---------------------------------------------------------------------------
__global__ void __launch_bounds__(256)
k_fc(const float* __restrict__ h, const float* __restrict__ fw,
     const float* __restrict__ fb, float* __restrict__ out) {
  const int nn = blockIdx.x, s = blockIdx.y;
  const int tid = threadIdx.x, lane = tid & 31, w = tid >> 5;
  __shared__ __attribute__((aligned(16))) _Float16 ht[128 * 136];  // [i][j], stride 136
  __shared__ __attribute__((aligned(16))) _Float16 wf[32 * 136];   // [o][j]
  const float* hb = h + ((size_t)nn * DOUT_ + s) * MM_ * MM_;
  for (int it = 0; it < 64; ++it) {
    int idx = tid + it * 256;
    int i = idx >> 7, j = idx & 127;
    ht[i * 136 + j] = (_Float16)hb[i * MM_ + j];
  }
  for (int it = 0; it < 16; ++it) {
    int idx = tid + it * 256;
    int o = idx >> 7, j = idx & 127;
    wf[o * 136 + j] = (_Float16)fw[o * MM_ + j];
  }
  __syncthreads();
  const int hi = lane >> 4, ln = lane & 15;
  const int I0 = w * 16;
  for (int ot = 0; ot < 2; ++ot) {
    v8f acc = {};
    for (int kc = 0; kc < 4; ++kc) {
      int jc = kc * 32;
      V16 a, b;
      const _Float16* pa = &ht[(I0 + ln) * 136 + jc + 16 * hi];  // A: m=i, k=j
      a.h[0] = *(const v8h*)pa;
      a.h[1] = *(const v8h*)(pa + 8);
      const _Float16* pb = &wf[(ot * 16 + ln) * 136 + jc + 16 * hi];  // B: k=j, n=o
      b.h[0] = *(const v8h*)pb;
      b.h[1] = *(const v8h*)(pb + 8);
      acc = __builtin_amdgcn_wmma_f32_16x16x32_f16(false, a.v, false, b.v, (short)0, acc, false, false);
    }
    const int o_ = ot * 16 + ln;
    const float fbv = fb[o_];
#pragma unroll
    for (int r = 0; r < 8; ++r) {
      int ig = I0 + r + 8 * hi;
      out[(((size_t)nn * DOUT_ + s) * MM_ + ig) * DOUT_ + o_] = acc[r] + fbv;
    }
  }
}

// ---------------------------------------------------------------------------
extern "C" void kernel_launch(void* const* d_in, const int* in_sizes, int n_in,
                              void* d_out, int out_size, void* d_ws, size_t ws_size,
                              hipStream_t stream) {
  const float* x   = (const float*)d_in[0];
  const float* c0  = (const float*)d_in[1];
  const float* b0  = (const float*)d_in[2];
  const float* c1  = (const float*)d_in[3];
  const float* b1  = (const float*)d_in[4];
  const float* db1 = (const float*)d_in[5];
  const float* c2  = (const float*)d_in[6];
  const float* b2  = (const float*)d_in[7];
  const float* db2 = (const float*)d_in[8];
  const float* fw  = (const float*)d_in[9];
  const float* fb  = (const float*)d_in[10];
  float* out = (float*)d_out;

  const size_t hElems = (size_t)NN_ * DOUT_ * MM_ * MM_;   // 8,388,608
  const size_t rElems = (size_t)NN_ * DOUT_ * 512;         // 262,144
  if (ws_size < (2 * hElems + 2 * rElems) * sizeof(float)) return;
  float* hA  = (float*)d_ws;
  float* hB  = hA + hElems;
  float* red = hB + hElems;
  float* mx  = red + rElems;

  k_eq1to2<<<dim3(NN_, DOUT_), 128, 0, stream>>>(x, c0, b0, hA);

  k_reduce<<<dim3(NN_, DOUT_), 128, 0, stream>>>(hA, red);
  k_mix<<<NN_, 128, 0, stream>>>(red, c1, mx);
  k_eq2to2_main<<<dim3(NN_, 8, 8), 256, 0, stream>>>(hA, c1, mx, b1, db1, hB);

  k_reduce<<<dim3(NN_, DOUT_), 128, 0, stream>>>(hB, red);
  k_mix<<<NN_, 128, 0, stream>>>(red, c2, mx);
  k_eq2to2_main<<<dim3(NN_, 8, 8), 256, 0, stream>>>(hB, c2, mx, b2, db2, hA);

  k_fc<<<dim3(NN_, DOUT_), 256, 0, stream>>>(hA, fw, fb, out);
}
